// Mamba2Mixer_56934086476138
// MI455X (gfx1250) — matrix operand order repaired and checked by reading, not compiled, not run on hardware
//
#include <hip/hip_runtime.h>
#include <hip/hip_bf16.h>
#include <math.h>

// ---------------- types for CDNA5 WMMA ----------------
typedef __bf16 bf16_t;
typedef __attribute__((ext_vector_type(16))) __bf16 v16bf;
typedef __attribute__((ext_vector_type(8)))  __bf16 v8bf;
typedef __attribute__((ext_vector_type(8)))  float  v8f;

#define B_   2
#define L_   4096
#define HID_ 768
#define I_   1536
#define NS_  128          // SSM state dim N
#define H_   24
#define P_   64
#define CS_  256          // chunk size
#define NC_  16           // number of chunks = L/CS
#define XBCW 1792         // I + 2N
#define ZW   3352         // 2*(I+N)+H
#define ROWS (B_*L_)      // 8192

union FragBF { v16bf v; v8bf h[2]; };

static __device__ __forceinline__ v8f v8f_zero() {
  v8f z;
  #pragma unroll
  for (int i = 0; i < 8; ++i) z[i] = 0.f;
  return z;
}

// Pack two floats -> two bf16 -> one 32-bit LDS store (v_cvt_pk_bf16_f32 + ds_store_b32).
// dst must be 4-byte aligned (all LDS strides here are even, pairs start at even k).
static __device__ __forceinline__ void store_pk(bf16_t* dst, float a, float b) {
  union { bf16_t h[2]; unsigned u; } p;
  p.h[0] = (bf16_t)a;
  p.h[1] = (bf16_t)b;
  *(unsigned*)dst = p.u;
}

// A fragment (16x32 bf16, M x K). smem: row-major [row][k], row stride = ldk elems.
// Lanes 0-15 row M=lane, K=0..15; lanes 16-31 same row, K=16..31 (matches B fragment).
static __device__ __forceinline__ v16bf load_a_frag(const bf16_t* smem, int lane, int ldk) {
  int row = lane & 15;
  int kb  = (lane < 16) ? 0 : 16;
  FragBF f;
  f.h[0] = *(const v8bf*)(smem + row * ldk + kb);
  f.h[1] = *(const v8bf*)(smem + row * ldk + kb + 8);
  return f.v;
}

// B fragment (32x16 bf16, K x N). smem holds B transposed: [n][k], row stride = ldk.
// Contiguous k order (matches A fragment): lanes 0-15 col N=lane K=0..15; lanes 16-31 K=16..31.
static __device__ __forceinline__ v16bf load_b_frag(const bf16_t* smem, int lane, int ldk) {
  int n  = lane & 15;
  int kb = (lane < 16) ? 0 : 16;
  FragBF f;
  f.h[0] = *(const v8bf*)(smem + n * ldk + kb);
  f.h[1] = *(const v8bf*)(smem + n * ldk + kb + 8);
  return f.v;
}

static __device__ __forceinline__ v8f wmma_bf16(v16bf a, v16bf b, v8f c) {
  return __builtin_amdgcn_wmma_f32_16x16x32_bf16(false, a, false, b, (short)0, c, false, false);
}

static __device__ __forceinline__ float silu(float x) {
  return x / (1.f + __expf(-x));
}

// Wave-parallel inclusive cumsum of Ah*s_dt[0..255] into s_cA (wave 0 only; caller syncs).
// Lane-local sums of 8 + 5-step shfl_up scan across the 32 lanes (wave32-native).
static __device__ __forceinline__ void chunk_scan(float Ah, const float* s_dt, float* s_cA, int tid) {
  if (tid < 32) {
    const int lane = tid;
    float loc[8];
    float seg = 0.f;
    #pragma unroll
    for (int i = 0; i < 8; ++i) { loc[i] = Ah * s_dt[lane * 8 + i]; seg += loc[i]; }
    float sc = seg;
    #pragma unroll
    for (int d = 1; d < 32; d <<= 1) {
      float nb = __shfl_up(sc, d, 32);
      if (lane >= d) sc += nb;
    }
    float run = sc - seg;   // exclusive prefix of this lane's segment
    #pragma unroll
    for (int i = 0; i < 8; ++i) { run += loc[i]; s_cA[lane * 8 + i] = run; }
  }
}

// ---------------- Kernel 1/7: bf16 WMMA GEMM, C = A(MxK) * B(KxN), fp32 in/out -------------
// Block: 256 threads (8 waves). Tile 128(M) x 128(N), K-step 32.
__global__ __launch_bounds__(256) void gemm_bf16(const float* __restrict__ A,
                                                 const float* __restrict__ Bm,
                                                 float* __restrict__ C,
                                                 int M, int N, int K) {
  __shared__ __align__(16) bf16_t As[128 * 40];   // [m][k], padded stride 40
  __shared__ __align__(16) bf16_t Bs[128 * 40];   // [n][k] (transposed), padded stride 40
  const int tid = threadIdx.x, lane = tid & 31, w = tid >> 5;
  const int m0 = blockIdx.y * 128, n0 = blockIdx.x * 128;
  const int wm = (w & 3) * 32, wn = (w >> 2) * 64;

  v8f acc[2][4];
  #pragma unroll
  for (int i = 0; i < 2; ++i)
    #pragma unroll
    for (int j = 0; j < 4; ++j) acc[i][j] = v8f_zero();

  const int nk = K >> 5;
  for (int kt = 0; kt < nk; ++kt) {
    const int k0 = kt << 5;
    // A tile: pairs along k -> global_load_b64 + packed bf16 ds_store_b32
    for (int i = tid; i < 128 * 16; i += 256) {
      int m = i >> 4, k2 = (i & 15) * 2;
      const float* ap = &A[(m0 + m) * K + k0 + k2];
      store_pk(&As[m * 40 + k2], ap[0], ap[1]);
    }
    // B tile (transposed into [n][k]): pairs along k (two strided global loads)
    for (int i = tid; i < 16 * 128; i += 256) {
      int k2 = (i >> 7) * 2, n = i & 127;
      bool in = (n0 + n) < N;
      float v0 = in ? Bm[(k0 + k2) * N + (n0 + n)] : 0.f;
      float v1 = in ? Bm[(k0 + k2 + 1) * N + (n0 + n)] : 0.f;
      store_pk(&Bs[n * 40 + k2], v0, v1);
    }
    __syncthreads();
    if (kt + 1 < nk)  // gfx1250 global_prefetch_b8 path for next A tile
      __builtin_prefetch(&A[(m0 + (tid >> 1)) * K + k0 + 32], 0, 0);

    v16bf af[2];
    #pragma unroll
    for (int i = 0; i < 2; ++i) af[i] = load_a_frag(&As[(wm + 16 * i) * 40], lane, 40);
    #pragma unroll
    for (int j = 0; j < 4; ++j) {
      v16bf bfg = load_b_frag(&Bs[(wn + 16 * j) * 40], lane, 40);
      #pragma unroll
      for (int i = 0; i < 2; ++i) acc[i][j] = wmma_bf16(af[i], bfg, acc[i][j]);
    }
    __syncthreads();
  }

  const int hi8 = (lane >> 4) << 3, ncol = lane & 15;
  #pragma unroll
  for (int i = 0; i < 2; ++i)
    #pragma unroll
    for (int j = 0; j < 4; ++j)
      #pragma unroll
      for (int e = 0; e < 8; ++e) {
        int row = m0 + wm + 16 * i + e + hi8;
        int col = n0 + wn + 16 * j + ncol;
        if (col < N) C[row * N + col] = acc[i][j][e];
      }
}

// ---------------- Kernel 2: causal depthwise conv (K=4) + bias + SiLU ----------------
__global__ __launch_bounds__(256) void conv_silu_kernel(const float* __restrict__ zx,
                                                        const float* __restrict__ cw,
                                                        const float* __restrict__ cb,
                                                        float* __restrict__ xconv) {
  const int col = blockIdx.x * 256 + threadIdx.x;   // 0..1791 (grid.x = 7)
  const int row = blockIdx.y;                        // 0..8191
  const int b = row >> 12, l = row & 4095;
  float a = cb[col];
  #pragma unroll
  for (int k = 0; k < 4; ++k) {
    int ls = l - 3 + k;
    if (ls >= 0) a += zx[(b * L_ + ls) * ZW + I_ + col] * cw[k * XBCW + col];
  }
  xconv[row * XBCW + col] = silu(a);
}

// ---------------- Kernel 2b: dt = softplus(dt + bias), clipped ----------------
__global__ __launch_bounds__(256) void dt_kernel(const float* __restrict__ zx,
                                                 const float* __restrict__ dt_bias,
                                                 float* __restrict__ dtw) {
  int idx = blockIdx.x * 256 + threadIdx.x;
  if (idx >= ROWS * H_) return;
  int row = idx / H_, h = idx % H_;
  float v = zx[row * ZW + (2 * I_ + 2 * NS_) + h] + dt_bias[h];
  float sp = (v > 20.f) ? v : log1pf(__expf(v));
  dtw[idx] = fmaxf(sp, 0.f);
}

// ---------------- Kernel 3: Y_diag (FlashAttention-style, per (b,c,h,stripe)) ----------
// Y_diag[l,p] = sum_s (C[l,:]·B[s,:]) * exp(cumA[l]-cumA[s]) * dt[s] * x[s,p],  s<=l
// Also writes Y_diag + D[h]*x into yout, and per-chunk total A sum (stripe 0).
__global__ __launch_bounds__(256) void ydiag_kernel(const float* __restrict__ xconv,
                                                    const float* __restrict__ dtw,
                                                    const float* __restrict__ A_log,
                                                    const float* __restrict__ Dv,
                                                    float* __restrict__ yout,
                                                    float* __restrict__ chunkAsum) {
  __shared__ float s_dt[CS_];
  __shared__ float s_cA[CS_];
  __shared__ __align__(16) bf16_t Qc[64 * 136];   // C stripe [l][d], d=0..127
  __shared__ __align__(16) bf16_t Bt[64 * 136];   // B tile   [s][d]
  __shared__ __align__(16) bf16_t xT[64 * 72];    // x tile transposed [p][s]
  __shared__ __align__(16) bf16_t Gm[64 * 72];    // masked G bf16 [l][s]

  const int tid = threadIdx.x, lane = tid & 31, w = tid >> 5;
  const int stripe = blockIdx.x;
  const int c = blockIdx.y / H_, h = blockIdx.y % H_, b = blockIdx.z;
  const int base = b * L_ + c * CS_;
  const float Ah = -__expf(A_log[h]);

  for (int t = tid; t < CS_; t += 256) s_dt[t] = dtw[(base + t) * H_ + h];
  __syncthreads();
  chunk_scan(Ah, s_dt, s_cA, tid);
  // C stripe: pairs along d
  for (int i = tid; i < 64 * 64; i += 256) {
    int l = i >> 6, d2 = (i & 63) * 2;
    const float* cp = &xconv[(base + stripe * 64 + l) * XBCW + (I_ + NS_) + d2];
    store_pk(&Qc[l * 136 + d2], cp[0], cp[1]);
  }
  __syncthreads();
  if (stripe == 0 && tid == 0) chunkAsum[(b * H_ + h) * NC_ + c] = s_cA[CS_ - 1];

  const int wm16 = (w & 3) * 16, wn32 = (w >> 2) * 32;
  const int hi8 = (lane >> 4) << 3, ncol = lane & 15;
  v8f accY[2]; accY[0] = v8f_zero(); accY[1] = v8f_zero();

  for (int st = 0; st <= stripe; ++st) {
    const int sb = st * 64;
    for (int i = tid; i < 64 * 64; i += 256) {
      int s = i >> 6, d2 = (i & 63) * 2;
      const float* bp = &xconv[(base + sb + s) * XBCW + I_ + d2];
      store_pk(&Bt[s * 136 + d2], bp[0], bp[1]);
    }
    // x tile transposed [p][s]: pair along s (two strided global loads per thread)
    for (int i = tid; i < 64 * 32; i += 256) {
      int sp = i >> 6, p = i & 63;
      int s2 = sp * 2;
      const float* xp = &xconv[(base + sb + s2) * XBCW + h * P_ + p];
      store_pk(&xT[p * 72 + s2], xp[0], xp[XBCW]);
    }
    __syncthreads();

    // G = C · B^T over d (K=128)
    v8f g[2]; g[0] = v8f_zero(); g[1] = v8f_zero();
    #pragma unroll
    for (int kt = 0; kt < 4; ++kt) {
      v16bf a = load_a_frag(&Qc[wm16 * 136 + kt * 32], lane, 136);
      #pragma unroll
      for (int j = 0; j < 2; ++j) {
        v16bf bb = load_b_frag(&Bt[(wn32 + 16 * j) * 136 + kt * 32], lane, 136);
        g[j] = wmma_bf16(a, bb, g[j]);
      }
    }
    // segsum mask * dt, write bf16 G
    #pragma unroll
    for (int j = 0; j < 2; ++j)
      #pragma unroll
      for (int e = 0; e < 8; ++e) {
        int ll = wm16 + e + hi8;
        int sl = wn32 + 16 * j + ncol;
        int lg = stripe * 64 + ll, sg = sb + sl;
        float gv = g[j][e];
        gv = (sg <= lg) ? gv * __expf(s_cA[lg] - s_cA[sg]) * s_dt[sg] : 0.f;
        Gm[ll * 72 + sl] = (bf16_t)gv;
      }
    __syncthreads();

    // Y += Gm · x  (K = s = 64)
    #pragma unroll
    for (int kt = 0; kt < 2; ++kt) {
      v16bf a2 = load_a_frag(&Gm[wm16 * 72 + kt * 32], lane, 72);
      #pragma unroll
      for (int j = 0; j < 2; ++j) {
        v16bf b2 = load_b_frag(&xT[(wn32 + 16 * j) * 72 + kt * 32], lane, 72);
        accY[j] = wmma_bf16(a2, b2, accY[j]);
      }
    }
    __syncthreads();
  }

  const float Dh = Dv[h];
  #pragma unroll
  for (int j = 0; j < 2; ++j)
    #pragma unroll
    for (int e = 0; e < 8; ++e) {
      int ll = wm16 + e + hi8;
      int p = wn32 + 16 * j + ncol;
      int row = base + stripe * 64 + ll;
      float xv = xconv[row * XBCW + h * P_ + p];
      yout[row * I_ + h * P_ + p] = accY[j][e] + Dh * xv;
    }
}

// ---------------- Kernel 3b: chunk end states S[d,p] = sum_l B[l,d]*decay[l]*dt[l]*x[l,p] ---
__global__ __launch_bounds__(256) void states_kernel(const float* __restrict__ xconv,
                                                     const float* __restrict__ dtw,
                                                     const float* __restrict__ A_log,
                                                     float* __restrict__ End) {
  __shared__ float s_dt[CS_];
  __shared__ float s_cA[CS_];
  __shared__ __align__(16) bf16_t BtT[128 * 72];   // B transposed [d][l]
  __shared__ __align__(16) bf16_t xsT[64 * 72];    // scaled x transposed [p][l]

  const int tid = threadIdx.x, lane = tid & 31, w = tid >> 5;
  const int c = blockIdx.x / H_, h = blockIdx.x % H_, b = blockIdx.y;
  const int base = b * L_ + c * CS_;
  const float Ah = -__expf(A_log[h]);

  for (int t = tid; t < CS_; t += 256) s_dt[t] = dtw[(base + t) * H_ + h];
  __syncthreads();
  chunk_scan(Ah, s_dt, s_cA, tid);
  __syncthreads();
  const float cAt = s_cA[CS_ - 1];

  v8f accS[4];
  #pragma unroll
  for (int j = 0; j < 4; ++j) accS[j] = v8f_zero();

  for (int lt = 0; lt < 4; ++lt) {
    // B transposed [d][l]: pair along l (two strided global loads per thread)
    for (int i = tid; i < 128 * 32; i += 256) {
      int lp = i >> 7, d = i & 127;
      int l2 = lp * 2;
      const float* bp = &xconv[(base + lt * 64 + l2) * XBCW + I_ + d];
      store_pk(&BtT[d * 72 + l2], bp[0], bp[XBCW]);
    }
    // scaled x transposed [p][l]
    for (int i = tid; i < 64 * 32; i += 256) {
      int lp = i >> 6, p = i & 63;
      int l2 = lp * 2;
      int lg = lt * 64 + l2;
      float sc0 = s_dt[lg] * __expf(cAt - s_cA[lg]);
      float sc1 = s_dt[lg + 1] * __expf(cAt - s_cA[lg + 1]);
      const float* xp = &xconv[(base + lg) * XBCW + h * P_ + p];
      store_pk(&xsT[p * 72 + l2], xp[0] * sc0, xp[XBCW] * sc1);
    }
    __syncthreads();
    #pragma unroll
    for (int kt = 0; kt < 2; ++kt) {
      v16bf a = load_a_frag(&BtT[(16 * w) * 72 + kt * 32], lane, 72);
      #pragma unroll
      for (int j = 0; j < 4; ++j) {
        v16bf bb = load_b_frag(&xsT[(16 * j) * 72 + kt * 32], lane, 72);
        accS[j] = wmma_bf16(a, bb, accS[j]);
      }
    }
    __syncthreads();
  }

  const int hi8 = (lane >> 4) << 3, ncol = lane & 15;
  float* Eg = End + (((size_t)(b * NC_ + c) * H_ + h) * P_ * NS_);   // [p][d]
  #pragma unroll
  for (int j = 0; j < 4; ++j)
    #pragma unroll
    for (int e = 0; e < 8; ++e) {
      int d = 16 * w + e + hi8;
      int p = 16 * j + ncol;
      Eg[p * NS_ + d] = accS[j][e];
    }
}

// ---------------- Kernel 4: inter-chunk recurrence, S_in written as bf16 [p][d] ----------
__global__ __launch_bounds__(256) void chain_kernel(const float* __restrict__ End,
                                                    const float* __restrict__ chunkAsum,
                                                    bf16_t* __restrict__ Sin) {
  const int h = blockIdx.x, b = blockIdx.y, tid = threadIdx.x;
  float Sr[32];
  #pragma unroll
  for (int i = 0; i < 32; ++i) Sr[i] = 0.f;
  for (int c = 0; c < NC_; ++c) {
    size_t off = ((size_t)(b * NC_ + c) * H_ + h) * (P_ * NS_);
    float ef = __expf(chunkAsum[(b * H_ + h) * NC_ + c]);
    #pragma unroll
    for (int i = 0; i < 32; ++i) {
      int idx = tid + i * 256;
      Sin[off + idx] = (bf16_t)Sr[i];          // state entering chunk c
      Sr[i] = Sr[i] * ef + End[off + idx];     // advance
    }
  }
}

// ---------------- Kernel 5: Y_off[l,p] = exp(cumA[l]) * sum_d C[l,d] * S_in[d,p] ---------
__global__ __launch_bounds__(256) void yoff_kernel(const float* __restrict__ xconv,
                                                   const float* __restrict__ dtw,
                                                   const float* __restrict__ A_log,
                                                   const bf16_t* __restrict__ Sin,
                                                   float* __restrict__ yout) {
  __shared__ float s_dt[CS_];
  __shared__ float s_cA[CS_];
  __shared__ __align__(16) bf16_t Cs[256 * 40];   // C chunk K-slab [l][d32]

  const int tid = threadIdx.x, lane = tid & 31, w = tid >> 5;
  const int c = blockIdx.x / H_, h = blockIdx.x % H_, b = blockIdx.y;
  const int base = b * L_ + c * CS_;
  const float Ah = -__expf(A_log[h]);

  for (int t = tid; t < CS_; t += 256) s_dt[t] = dtw[(base + t) * H_ + h];
  __syncthreads();
  chunk_scan(Ah, s_dt, s_cA, tid);

  const int wm = 32 * w;
  const int hi8 = (lane >> 4) << 3, ncol = lane & 15;
  v8f acc[2][4];
  #pragma unroll
  for (int i = 0; i < 2; ++i)
    #pragma unroll
    for (int j = 0; j < 4; ++j) acc[i][j] = v8f_zero();

  const bf16_t* Sg = Sin + (((size_t)(b * NC_ + c) * H_ + h) * P_ * NS_);  // [p][d]

  for (int ktd = 0; ktd < 4; ++ktd) {
    for (int i = tid; i < 256 * 16; i += 256) {
      int l = i >> 4, d2 = (i & 15) * 2;
      const float* cp = &xconv[(base + l) * XBCW + (I_ + NS_) + ktd * 32 + d2];
      store_pk(&Cs[l * 40 + d2], cp[0], cp[1]);
    }
    __syncthreads();
    v16bf a[2];
    #pragma unroll
    for (int i = 0; i < 2; ++i) a[i] = load_a_frag(&Cs[(wm + 16 * i) * 40], lane, 40);
    #pragma unroll
    for (int j = 0; j < 4; ++j) {
      int prow = 16 * j + ncol;
      int kb = ktd * 32 + ((lane < 16) ? 0 : 16);
      FragBF f;   // B-fragment (k=d, n=p) straight from global bf16 S_in (16B loads)
      f.h[0] = *(const v8bf*)(Sg + prow * NS_ + kb);
      f.h[1] = *(const v8bf*)(Sg + prow * NS_ + kb + 8);
      #pragma unroll
      for (int i = 0; i < 2; ++i) acc[i][j] = wmma_bf16(a[i], f.v, acc[i][j]);
    }
    __syncthreads();
  }

  #pragma unroll
  for (int i = 0; i < 2; ++i)
    #pragma unroll
    for (int j = 0; j < 4; ++j)
      #pragma unroll
      for (int e = 0; e < 8; ++e) {
        int ll = wm + 16 * i + e + hi8;
        int p = 16 * j + ncol;
        int row = base + ll;
        yout[row * I_ + h * P_ + p] += acc[i][j][e] * __expf(s_cA[ll]);
      }
}

// ---------------- Kernel 6: gated RMSNorm in place ----------------
__global__ __launch_bounds__(256) void rmsnorm_kernel(float* __restrict__ yout,
                                                      const float* __restrict__ zx,
                                                      const float* __restrict__ norm_w) {
  __shared__ float red[256];
  const int row = blockIdx.x, tid = threadIdx.x;
  float t[6];
  float part = 0.f;
  #pragma unroll
  for (int k = 0; k < 6; ++k) {
    int i = tid + k * 256;
    float z = zx[row * ZW + i];
    float tv = yout[row * I_ + i] * silu(z);
    t[k] = tv;
    part += tv * tv;
  }
  red[tid] = part;
  __syncthreads();
  for (int s = 128; s > 0; s >>= 1) {
    if (tid < s) red[tid] += red[tid + s];
    __syncthreads();
  }
  float rs = rsqrtf(red[0] / (float)I_ + 1e-5f);
  #pragma unroll
  for (int k = 0; k < 6; ++k) {
    int i = tid + k * 256;
    yout[row * I_ + i] = t[k] * rs * norm_w[i];
  }
}

// ---------------- host launcher ----------------
extern "C" void kernel_launch(void* const* d_in, const int* in_sizes, int n_in,
                              void* d_out, int out_size, void* d_ws, size_t ws_size,
                              hipStream_t stream) {
  (void)in_sizes; (void)n_in; (void)out_size; (void)ws_size;
  const float* hs       = (const float*)d_in[0];  // (2,4096,768)
  const float* in_proj  = (const float*)d_in[1];  // (768,3352)
  const float* conv_w   = (const float*)d_in[2];  // (4,1792)
  const float* conv_b   = (const float*)d_in[3];  // (1792,)
  const float* dt_bias  = (const float*)d_in[4];  // (24,)
  const float* A_log    = (const float*)d_in[5];  // (24,)
  const float* Dv       = (const float*)d_in[6];  // (24,)
  const float* norm_w   = (const float*)d_in[7];  // (1536,)
  const float* out_proj = (const float*)d_in[8];  // (1536,768)

  char* ws = (char*)d_ws;
  size_t off = 0;
  auto carve = [&](size_t bytes) { size_t o = off; off = (off + bytes + 255) & ~(size_t)255; return o; };
  float*  zx       = (float*)(ws + carve((size_t)ROWS * ZW * 4));          // 109.8 MB
  float*  xconv    = (float*)(ws + carve((size_t)ROWS * XBCW * 4));        //  58.7 MB
  float*  dtw      = (float*)(ws + carve((size_t)ROWS * H_ * 4));          //   0.8 MB
  float*  yout     = (float*)(ws + carve((size_t)ROWS * I_ * 4));          //  50.3 MB
  float*  End      = (float*)(ws + carve((size_t)B_ * NC_ * H_ * P_ * NS_ * 4)); // 25.2 MB
  bf16_t* Sin      = (bf16_t*)(ws + carve((size_t)B_ * NC_ * H_ * P_ * NS_ * 2)); // 12.6 MB
  float*  chunkA   = (float*)(ws + carve((size_t)B_ * H_ * NC_ * 4));      // tiny
  // total ~258 MB

  // 1) zxbcdt = hidden @ in_proj_w
  gemm_bf16<<<dim3((ZW + 127) / 128, ROWS / 128), 256, 0, stream>>>(hs, in_proj, zx, ROWS, ZW, HID_);
  // 2) conv + SiLU ; dt softplus
  conv_silu_kernel<<<dim3(XBCW / 256, ROWS), 256, 0, stream>>>(zx, conv_w, conv_b, xconv);
  dt_kernel<<<dim3((ROWS * H_ + 255) / 256), 256, 0, stream>>>(zx, dt_bias, dtw);
  // 3) Y_diag + D*x, chunk A sums
  ydiag_kernel<<<dim3(4, NC_ * H_, B_), 256, 0, stream>>>(xconv, dtw, A_log, Dv, yout, chunkA);
  // 3b) per-chunk end states
  states_kernel<<<dim3(NC_ * H_, B_), 256, 0, stream>>>(xconv, dtw, A_log, End);
  // 4) inter-chunk recurrence -> S_in (bf16)
  chain_kernel<<<dim3(H_, B_), 256, 0, stream>>>(End, chunkA, Sin);
  // 5) Y_off accumulate
  yoff_kernel<<<dim3(NC_ * H_, B_), 256, 0, stream>>>(xconv, dtw, A_log, Sin, yout);
  // 6) gated RMSNorm (in place)
  rmsnorm_kernel<<<dim3(ROWS), 256, 0, stream>>>(yout, zx, norm_w);
  // 7) out = y @ out_proj_w
  gemm_bf16<<<dim3(HID_ / 128, ROWS / 128), 256, 0, stream>>>(yout, out_proj, (float*)d_out, ROWS, HID_, I_);
}
